// FilterInitializerLinear_68478958567838
// MI455X (gfx1250) — matrix operand order repaired and checked by reading, not compiled, not run on hardware
//
#include <hip/hip_runtime.h>

// ---------------------------------------------------------------------------
// Problem constants (from the reference)
// ---------------------------------------------------------------------------
#define NI    3
#define NS    64
#define NIMG  (NI * NS)          // 192 conv batches
#define CCH   512                // channels in == channels out
#define HIMG  22
#define WIMG  22
#define NPIX  (HIMG * WIMG)      // 484
#define POOL  4

// GEMM view of the conv: M = NIMG*NPIX, N = CCH, K = 9*CCH
// Per-wave tile: 32(M) x 64(N); K stepped 32 at a time via wmma bf16 16x16x32.

typedef float          v8f    __attribute__((ext_vector_type(8)));
typedef __bf16         v16bf  __attribute__((ext_vector_type(16)));
typedef unsigned short u16x8  __attribute__((ext_vector_type(8)));
typedef unsigned short u16x16 __attribute__((ext_vector_type(16)));

__device__ __forceinline__ unsigned short f2bf(float f) {
  unsigned int u = __float_as_uint(f);
  u += 0x7FFFu + ((u >> 16) & 1u);          // round-to-nearest-even
  return (unsigned short)(u >> 16);
}
__device__ __forceinline__ float bf2f(unsigned short h) {
  return __uint_as_float(((unsigned int)h) << 16);
}
__device__ __forceinline__ v16bf cat16(u16x8 lo, u16x8 hi) {
  u16x16 t = __builtin_shufflevector(lo, hi, 0,1,2,3,4,5,6,7,8,9,10,11,12,13,14,15);
  return __builtin_bit_cast(v16bf, t);
}

// ---------------------------------------------------------------------------
// Stage 0a: feat NCHW f32 -> NHWC bf16 (LDS tile transpose, coalesced r/w)
// grid (16, 16, 192), block 256
// ---------------------------------------------------------------------------
__global__ __launch_bounds__(256) void feat_nchw_to_nhwc_bf16(
    const float* __restrict__ in, unsigned short* __restrict__ outb) {
  __shared__ float tile[32][33];
  const int n  = blockIdx.z;
  const int p0 = blockIdx.x * 32;
  const int c0 = blockIdx.y * 32;
  const int t  = threadIdx.x;
  const int pp = t & 31;
  const int cw = t >> 5;                      // 0..7
#pragma unroll
  for (int k = 0; k < 4; ++k) {
    int c = c0 + cw + k * 8;
    int p = p0 + pp;
    float v = 0.f;
    if (p < NPIX) v = in[((size_t)n * CCH + c) * NPIX + p];
    tile[cw + k * 8][pp] = v;
  }
  __syncthreads();
  const int cc = t & 31;
  const int pw = t >> 5;
#pragma unroll
  for (int k = 0; k < 4; ++k) {
    int p = p0 + pw + k * 8;
    if (p < NPIX)
      outb[((size_t)n * NPIX + p) * CCH + c0 + cc] = f2bf(tile[cc][pw + k * 8]);
  }
}

// ---------------------------------------------------------------------------
// Stage 0b: conv_w (co,ci,3,3) f32 -> B-fragment-order bf16.
// Layout: Bg[kk][kc][ct][lane][e]; lane<16: N=lane, K=e; lane>=16: N=lane-16,
// K=16+e (contiguous k order, matching the A loads).  K-dim = kk*512 + ci.
// Total elements: 9*16*32*32*16 = 2359296.  grid 9216, block 256.
// ---------------------------------------------------------------------------
__global__ __launch_bounds__(256) void conv_w_to_frags(
    const float* __restrict__ w, unsigned short* __restrict__ Bg) {
  int id = blockIdx.x * 256 + threadIdx.x;
  int e  = id & 15;
  int l  = (id >> 4) & 31;
  int ct = (id >> 9) & 31;
  int kc = (id >> 14) & 15;
  int kk = id >> 18;                          // 0..8
  int co = ct * 16 + (l & 15);
  int ci = kc * 32 + (l >> 4) * 16 + e;
  Bg[id] = f2bf(w[((size_t)co * CCH + ci) * 9 + kk]);
}

// ---------------------------------------------------------------------------
// Zero pad (1KB) used as the A source for out-of-image / out-of-range rows,
// so the GEMM inner loop has no divergent control flow.
// ---------------------------------------------------------------------------
__global__ __launch_bounds__(256) void zero_pad_kernel(unsigned short* p) {
  p[blockIdx.x * 256 + threadIdx.x] = 0;
}

// ---------------------------------------------------------------------------
// Stage 1: implicit-GEMM 3x3 conv via v_wmma_f32_16x16x32_bf16.
// Wave tile 32(M px) x 64(N co); block = 4 waves = 4 consecutive M tiles.
// Software-pipelined K loop (load kc+1 while computing kc), branch-free A.
// grid (768, 8), block 128.
// ---------------------------------------------------------------------------
__global__ __launch_bounds__(128) void conv3x3_wmma_bf16(
    const unsigned short* __restrict__ featb,   // [192][484][512] bf16 NHWC
    const unsigned short* __restrict__ Bg,      // weight fragments
    const unsigned short* __restrict__ zpad,    // 512 zero bf16
    const float*          __restrict__ bias,    // [512]
    unsigned short*       __restrict__ outb)    // [192][484][512] bf16 NHWC
{
  const int lane = threadIdx.x & 31;
  const int wave = threadIdx.x >> 5;
  const int mt   = blockIdx.x * 4 + wave;       // 0..3071
  const int n    = mt >> 4;                     // image (192)
  const int p0   = (mt & 15) << 5;              // first pixel of 32-px tile
  const int half = lane >> 4;
  const int mloc = lane & 15;
  const int by4  = blockIdx.y * 4;              // base co-tile (of 16)

  v8f acc[2][4] = {};

  const size_t imgBase = (size_t)n * NPIX * CCH;

#pragma unroll 1
  for (int kk = 0; kk < 9; ++kk) {
    const int dh = kk / 3 - 1, dw = kk % 3 - 1;
    // Branch-free A base pointers: invalid lanes read the zero pad.
    const unsigned short* aBase[2];
#pragma unroll
    for (int mi = 0; mi < 2; ++mi) {
      int p = p0 + mi * 16 + mloc;
      int h = p / WIMG + dh;
      int w = p % WIMG + dw;
      bool ok = (p < NPIX) & (h >= 0) & (h < HIMG) & (w >= 0) & (w < WIMG);
      int off = ok ? (h * WIMG + w) : 0;
      const unsigned short* rp = ok ? (featb + imgBase + (size_t)off * CCH) : zpad;
      aBase[mi] = rp + half * 16;
    }
    const unsigned short* bPtr =
        Bg + (size_t)(kk * 16) * 16384 + (size_t)by4 * 512 + lane * 16;

    // Warm L2/WGP$ for the next tap's B fragments (global_prefetch_b8).
    if (kk < 8) __builtin_prefetch(bPtr + (size_t)16 * 16384, 0, 1);

    // Preload kc = 0 fragments.
    v16bf aCur[2], bCur[4];
#pragma unroll
    for (int mi = 0; mi < 2; ++mi)
      aCur[mi] = cat16(*(const u16x8*)(aBase[mi]),
                       *(const u16x8*)(aBase[mi] + 8));
#pragma unroll
    for (int ni = 0; ni < 4; ++ni)
      bCur[ni] = cat16(*(const u16x8*)(bPtr + ni * 512),
                       *(const u16x8*)(bPtr + ni * 512 + 8));

#pragma unroll 2
    for (int kc = 0; kc < 16; ++kc) {
      const int kcn = (kc + 1) & 15;            // wrap: last preload is benign
      v16bf aNxt[2], bNxt[4];
#pragma unroll
      for (int mi = 0; mi < 2; ++mi)
        aNxt[mi] = cat16(*(const u16x8*)(aBase[mi] + kcn * 32),
                         *(const u16x8*)(aBase[mi] + kcn * 32 + 8));
      const unsigned short* bp = bPtr + (size_t)kcn * 16384;
#pragma unroll
      for (int ni = 0; ni < 4; ++ni)
        bNxt[ni] = cat16(*(const u16x8*)(bp + ni * 512),
                         *(const u16x8*)(bp + ni * 512 + 8));
#pragma unroll
      for (int ni = 0; ni < 4; ++ni)
#pragma unroll
        for (int mi = 0; mi < 2; ++mi)
          acc[mi][ni] = __builtin_amdgcn_wmma_f32_16x16x32_bf16(
              false, aCur[mi], false, bCur[ni], (short)0, acc[mi][ni],
              false, false);
#pragma unroll
      for (int mi = 0; mi < 2; ++mi) aCur[mi] = aNxt[mi];
#pragma unroll
      for (int ni = 0; ni < 4; ++ni) bCur[ni] = bNxt[ni];
    }
  }

  // C/D layout: lane<16 -> M = v, N = lane; lane>=16 -> M = 8+v, N = lane-16.
#pragma unroll
  for (int mi = 0; mi < 2; ++mi) {
#pragma unroll
    for (int v = 0; v < 8; ++v) {
      int p = p0 + mi * 16 + half * 8 + v;
      if (p < NPIX) {
#pragma unroll
        for (int ni = 0; ni < 4; ++ni) {
          int co = (by4 + ni) * 16 + mloc;
          float val = acc[mi][ni][v] + bias[co];
          outb[imgBase + (size_t)p * CCH + co] = f2bf(val);
        }
      }
    }
  }
}

// ---------------------------------------------------------------------------
// Stage 2: PrRoI pool 4x4 + mean over images + filter norm.
// Closed-form per-gridpoint integration weight (== reference scatter-add).
// grid 64 (one per sequence), block 256 (each thread: 2 channels).
// ---------------------------------------------------------------------------
__device__ __forceinline__ float axis_w(float s, float e, float fi) {
  // cell [fi, fi+1]: (1-t) part
  float a1 = fmaxf(s, fi);
  float a2 = fmaxf(fminf(e, fi + 1.f), a1);
  float al = a1 - fi, lm = a2 - fi;
  float f0 = (lm - 0.5f * lm * lm) - (al - 0.5f * al * al);
  // cell [fi-1, fi]: t part
  float fj = fi - 1.f;
  float b1 = fmaxf(s, fj);
  float b2 = fmaxf(fminf(e, fj + 1.f), b1);
  float bl = b1 - fj, l2 = b2 - fj;
  float f1 = 0.5f * (l2 * l2 - bl * bl);
  return f0 + f1;
}

__global__ __launch_bounds__(256) void prroi_mean_norm(
    const unsigned short* __restrict__ conv,   // [192][484][512] bf16 NHWC
    const float*          __restrict__ bb,     // [3][64][4] (x,y,w,h)
    float*                __restrict__ out)    // [64][512][4][4] f32
{
  const int ns = blockIdx.x;
  const int t  = threadIdx.x;
  __shared__ float sWy[POOL][HIMG];
  __shared__ float sWx[POOL][WIMG];
  __shared__ float sInvArea;
  float acc[2][16];
#pragma unroll
  for (int i = 0; i < 2; ++i)
#pragma unroll
    for (int j = 0; j < 16; ++j) acc[i][j] = 0.f;

  for (int ni = 0; ni < NI; ++ni) {
    const int n = ni * NS + ns;
    if (t < 2 * POOL * HIMG) {                 // 176 threads build weights
      const float* r = bb + (size_t)n * 4;
      const float sc = 1.f / 16.f;
      float x1 = r[0] * sc, y1 = r[1] * sc;
      float x2 = (r[0] + r[2]) * sc, y2 = (r[1] + r[3]) * sc;
      float bw = (x2 - x1) * 0.25f, bh = (y2 - y1) * 0.25f;
      int isY = (t < POOL * HIMG);
      int tt  = isY ? t : (t - POOL * HIMG);
      int p = tt / HIMG, i = tt % HIMG;
      float s = isY ? (y1 + p * bh) : (x1 + p * bw);
      float e = s + (isY ? bh : bw);
      float w = axis_w(s, e, (float)i);
      if (isY) sWy[p][i] = w; else sWx[p][i] = w;
      if (t == 0) {
        float area = bw * bh;
        sInvArea = (area > 0.f) ? (1.f / area) : 0.f;
      }
    }
    __syncthreads();
    const float invA = sInvArea;
    float part[2][16];
#pragma unroll
    for (int i = 0; i < 2; ++i)
#pragma unroll
      for (int j = 0; j < 16; ++j) part[i][j] = 0.f;

    const unsigned short* base = conv + (size_t)n * NPIX * CCH + t;
    for (int h = 0; h < HIMG; ++h) {
      float wy[POOL];
#pragma unroll
      for (int py = 0; py < POOL; ++py) wy[py] = sWy[py][h];
      if (wy[0] == 0.f && wy[1] == 0.f && wy[2] == 0.f && wy[3] == 0.f)
        continue;                               // uniform branch (shared data)
      for (int w2 = 0; w2 < WIMG; ++w2) {
        float wx[POOL];
#pragma unroll
        for (int px = 0; px < POOL; ++px) wx[px] = sWx[px][w2];
        const unsigned short* xp = base + (size_t)(h * WIMG + w2) * CCH;
        float v0 = bf2f(xp[0]);
        float v1 = bf2f(xp[256]);
#pragma unroll
        for (int py = 0; py < POOL; ++py)
#pragma unroll
          for (int px = 0; px < POOL; ++px) {
            float wgt = wy[py] * wx[px];
            part[0][py * 4 + px] = fmaf(wgt, v0, part[0][py * 4 + px]);
            part[1][py * 4 + px] = fmaf(wgt, v1, part[1][py * 4 + px]);
          }
      }
    }
#pragma unroll
    for (int i = 0; i < 2; ++i)
#pragma unroll
      for (int j = 0; j < 16; ++j) acc[i][j] = fmaf(part[i][j], invA, acc[i][j]);
    __syncthreads();
  }

  const float norm = (1.f / (float)NI) / ((float)CCH * 16.f);
#pragma unroll
  for (int cc = 0; cc < 2; ++cc) {
    int c = t + cc * 256;
#pragma unroll
    for (int b = 0; b < 16; ++b)
      out[((size_t)ns * CCH + c) * 16 + b] = acc[cc][b] * norm;
  }
}

// ---------------------------------------------------------------------------
// Workspace layout (bytes)
// ---------------------------------------------------------------------------
static const size_t FEATB_BYTES = (size_t)NIMG * NPIX * CCH * 2;   // 95,158,272
static const size_t BG_BYTES    = (size_t)9 * 16 * 32 * 512 * 2;   //  4,718,592
static const size_t FEATB_OFF   = 0;
static const size_t BG_OFF      = FEATB_OFF + FEATB_BYTES;
static const size_t CONV_OFF    = BG_OFF + BG_BYTES;
static const size_t ZPAD_OFF    = CONV_OFF + FEATB_BYTES;          // 1KB zeros

extern "C" void kernel_launch(void* const* d_in, const int* in_sizes, int n_in,
                              void* d_out, int out_size, void* d_ws, size_t ws_size,
                              hipStream_t stream) {
  (void)in_sizes; (void)n_in; (void)out_size; (void)ws_size;
  const float* feat   = (const float*)d_in[0];   // (3,64,512,22,22)
  const float* bb     = (const float*)d_in[1];   // (3,64,4)
  const float* conv_w = (const float*)d_in[2];   // (512,512,3,3)
  const float* conv_b = (const float*)d_in[3];   // (512)
  float* out = (float*)d_out;                    // (64,512,4,4)

  char* ws = (char*)d_ws;
  unsigned short* featb   = (unsigned short*)(ws + FEATB_OFF);
  unsigned short* Bg      = (unsigned short*)(ws + BG_OFF);
  unsigned short* convout = (unsigned short*)(ws + CONV_OFF);
  unsigned short* zpad    = (unsigned short*)(ws + ZPAD_OFF);

  dim3 gT(16, 16, NIMG);
  feat_nchw_to_nhwc_bf16<<<gT, 256, 0, stream>>>(feat, featb);

  conv_w_to_frags<<<9216, 256, 0, stream>>>(conv_w, Bg);

  zero_pad_kernel<<<2, 256, 0, stream>>>(zpad);

  dim3 gC(768, 8);
  conv3x3_wmma_bf16<<<gC, 128, 0, stream>>>(featb, Bg, zpad, conv_b, convout);

  prroi_mean_norm<<<NS, 256, 0, stream>>>(convout, bb, out);
}
